// IntegratedChromaTransformer2DModel_39367670235502
// MI455X (gfx1250) — matrix operand order repaired and checked by reading, not compiled, not run on hardware
//
#include <hip/hip_runtime.h>
#include <hip/hip_bf16.h>
#include <math.h>

// ---------------------------------------------------------------------------
// Chroma transformer forward (B=1) for MI455X / gfx1250 (wave32, WMMA).
// All large GEMMs run on v_wmma_f32_16x16x32_bf16: fp32 operands are read
// from HBM once and converted to bf16 while staging into LDS; accumulation
// stays fp32.  GEMM main loop is double-buffered in LDS: global loads for
// tile k+1 are issued, WMMAs run on tile k, then cvt+ds_store fill the
// alternate buffer -> HBM latency hides behind matrix issue.
// Attention = batched WMMA scores -> fp32 softmax -> WMMA P@V.
// ---------------------------------------------------------------------------

typedef __bf16 bf16_t;
typedef bf16_t bf16x8  __attribute__((ext_vector_type(8)));
typedef bf16_t bf16x16 __attribute__((ext_vector_type(16)));
typedef float  f32x8   __attribute__((ext_vector_type(8)));

#define HIDDEN   3072
#define HEADS    24
#define HEAD_D   128
#define LTXT     512
#define LIMG     1024
#define LALL     1536
#define MLPH     12288
#define MODL     17
#define APPRH    5120
#define CTXD     4096
#define INCH     64
#define F16MAXV  65504.0f

__device__ __forceinline__ float gelu_tanh_f(float x) {
    float x3 = x * x * x;
    return 0.5f * x * (1.0f + tanhf(0.7978845608028654f * (x + 0.044715f * x3)));
}
__device__ __forceinline__ float silu_f(float x) { return x / (1.0f + __expf(-x)); }

// A fragment: lane<16 -> M=lane, K in {kb..kb+7, kb+16..kb+23} with kb=0;
// lane>=16 -> kb=8.  (ISA 7.12.2, 16-bit A 16x32 layout.)
__device__ __forceinline__ bf16x16 ldfragA(const bf16_t* base, int row, int lane) {
    int r  = row + (lane & 15);
    int kb = (lane < 16) ? 0 : 8;
    const bf16_t* p = base + r * 40 + kb;          // A-tile LDS stride = 40 bf16 (80B)
    bf16x8 lo = *reinterpret_cast<const bf16x8*>(p);
    bf16x8 hi = *reinterpret_cast<const bf16x8*>(p + 16);
    return __builtin_shufflevector(lo, hi, 0,1,2,3,4,5,6,7,8,9,10,11,12,13,14,15);
}
// B fragment: lane -> N = lane&15, K in {kb..kb+7, kb+16..kb+23}, kb = 8*(lane>=16).
// LDS B tile is stored [K=32][N=64] with stride 80 bf16 (160B, 16B aligned rows).
__device__ __forceinline__ bf16x16 ldfragB(const bf16_t* base, int ncol, int lane) {
    const bf16_t* p = base + ncol + (lane & 15);
    int kb = (lane < 16) ? 0 : 8;
    bf16x16 v;
#pragma unroll
    for (int s = 0; s < 8; ++s) {
        v[s]     = p[(kb + s) * 80];
        v[s + 8] = p[(kb + 16 + s) * 80];
    }
    return v;
}

// Generic batched GEMM: C[b] = epilogue(alpha * A[b] (MxK) * op(B[b]) + bias)
//   bNN == 0: B is [N,K] row-major (x @ W^T);  bNN == 1: B is [K,N] row-major.
//   act: 0 none, 1 gelu-tanh, 2 silu.  res/gate: out = res + gate[n]*val.
__global__ __launch_bounds__(256) void gemm_wmma(
    const float* __restrict__ A, int lda, long long sA,
    const float* __restrict__ B, int ldb, long long sB, int bNN,
    float* __restrict__ C, int ldc, long long sC,
    int M, int N, int K,
    const float* __restrict__ bias, float alpha, int act,
    const float* __restrict__ res, const float* __restrict__ gate, int clampOut)
{
    __shared__ __align__(16) bf16_t As[2][128][40];   // [M=128][K=32] (+pad)
    __shared__ __align__(16) bf16_t Bs[2][32][80];    // [K=32][N=64]  (+pad)

    const int tid  = threadIdx.x;
    const int lane = tid & 31;
    const int wid  = tid >> 5;
    const int wm   = (wid & 3) * 32;   // wave M offset inside block tile
    const int wn   = (wid >> 2) * 32;  // wave N offset inside block tile
    const int bm   = blockIdx.y * 128;
    const int bn   = blockIdx.x * 64;

    A += (size_t)blockIdx.z * sA;
    B += (size_t)blockIdx.z * sB;
    C += (size_t)blockIdx.z * sC;

    f32x8 acc[2][2] = {};

    // Phase 1: issue all global b128 loads for tile k0 into registers.
    auto loadTiles = [&](int k0, float4* ra, float4* rb) {
        int colA = (tid & 7) * 4;
#pragma unroll
        for (int i = 0; i < 4; ++i) {
            int gm = bm + (tid >> 3) + i * 32;
            ra[i] = make_float4(0.f, 0.f, 0.f, 0.f);
            if (gm < M) {
                const float* ap = A + (size_t)gm * lda + k0 + colA;
                ra[i] = *reinterpret_cast<const float4*>(ap);
                if (k0 + 32 < K) __builtin_prefetch(ap + 32, 0, 3);  // near-scope prefetch
            }
        }
        if (!bNN) {
#pragma unroll
            for (int i = 0; i < 2; ++i) {
                int gn = bn + (tid >> 3) + i * 32;
                rb[i] = make_float4(0.f, 0.f, 0.f, 0.f);
                if (gn < N) {
                    const float* bp = B + (size_t)gn * ldb + k0 + colA;
                    rb[i] = *reinterpret_cast<const float4*>(bp);
                    if (k0 + 32 < K) __builtin_prefetch(bp + 32, 0, 3);
                }
            }
        } else {
            int n4 = (tid & 15) * 4;
#pragma unroll
            for (int i = 0; i < 2; ++i) {
                int kk = (tid >> 4) + i * 16;
                rb[i] = make_float4(0.f, 0.f, 0.f, 0.f);
                if (bn + n4 < N)
                    rb[i] = *reinterpret_cast<const float4*>(B + (size_t)(k0 + kk) * ldb + bn + n4);
            }
        }
    };
    // Phase 2: fp32 -> bf16 convert and ds_store into LDS buffer `buf`.
    auto storeTiles = [&](int buf, const float4* ra, const float4* rb) {
        int colA = (tid & 7) * 4;
#pragma unroll
        for (int i = 0; i < 4; ++i) {
            int row = (tid >> 3) + i * 32;
            As[buf][row][colA + 0] = (bf16_t)ra[i].x;
            As[buf][row][colA + 1] = (bf16_t)ra[i].y;
            As[buf][row][colA + 2] = (bf16_t)ra[i].z;
            As[buf][row][colA + 3] = (bf16_t)ra[i].w;
        }
        if (!bNN) {        // weights [N,K]: transpose into Bs[k][n]
#pragma unroll
            for (int i = 0; i < 2; ++i) {
                int row = (tid >> 3) + i * 32;
                Bs[buf][colA + 0][row] = (bf16_t)rb[i].x;
                Bs[buf][colA + 1][row] = (bf16_t)rb[i].y;
                Bs[buf][colA + 2][row] = (bf16_t)rb[i].z;
                Bs[buf][colA + 3][row] = (bf16_t)rb[i].w;
            }
        } else {           // B [K,N]: direct copy into Bs[k][n]
            int n4 = (tid & 15) * 4;
#pragma unroll
            for (int i = 0; i < 2; ++i) {
                int kk = (tid >> 4) + i * 16;
                Bs[buf][kk][n4 + 0] = (bf16_t)rb[i].x;
                Bs[buf][kk][n4 + 1] = (bf16_t)rb[i].y;
                Bs[buf][kk][n4 + 2] = (bf16_t)rb[i].z;
                Bs[buf][kk][n4 + 3] = (bf16_t)rb[i].w;
            }
        }
    };

    float4 ra[4], rb[2];
    loadTiles(0, ra, rb);
    storeTiles(0, ra, rb);
    __syncthreads();

    const int nk = K >> 5;
    for (int kt = 0; kt < nk; ++kt) {
        const int cur = kt & 1;
        const bool hasNext = (kt + 1) < nk;
        if (hasNext) loadTiles((kt + 1) << 5, ra, rb);   // loads in flight over WMMAs

        bf16x16 a0 = ldfragA(&As[cur][0][0], wm + 0,  lane);
        bf16x16 a1 = ldfragA(&As[cur][0][0], wm + 16, lane);
        bf16x16 b0 = ldfragB(&Bs[cur][0][0], wn + 0,  lane);
        bf16x16 b1 = ldfragB(&Bs[cur][0][0], wn + 16, lane);
        acc[0][0] = __builtin_amdgcn_wmma_f32_16x16x32_bf16(false, a0, false, b0, (short)0, acc[0][0], false, false);
        acc[0][1] = __builtin_amdgcn_wmma_f32_16x16x32_bf16(false, a0, false, b1, (short)0, acc[0][1], false, false);
        acc[1][0] = __builtin_amdgcn_wmma_f32_16x16x32_bf16(false, a1, false, b0, (short)0, acc[1][0], false, false);
        acc[1][1] = __builtin_amdgcn_wmma_f32_16x16x32_bf16(false, a1, false, b1, (short)0, acc[1][1], false, false);

        if (hasNext) storeTiles(cur ^ 1, ra, rb);
        __syncthreads();
    }

    // ---- epilogue: C/D layout m = vgpr + 8*(lane>=16), n = lane&15 --------
#pragma unroll
    for (int tm = 0; tm < 2; ++tm) {
#pragma unroll
        for (int tn = 0; tn < 2; ++tn) {
            int mBase = bm + wm + tm * 16 + ((lane >> 4) << 3);
            int n     = bn + wn + tn * 16 + (lane & 15);
            if (n >= N) continue;
#pragma unroll
            for (int e = 0; e < 8; ++e) {
                int m = mBase + e;
                if (m >= M) continue;
                float x = alpha * acc[tm][tn][e];
                if (bias) x += bias[n];
                if (act == 1)      x = gelu_tanh_f(x);
                else if (act == 2) x = silu_f(x);
                if (res) {
                    float g = gate ? gate[n] : 1.0f;
                    x = res[(size_t)m * ldc + n] + g * x;
                }
                if (clampOut) x = fminf(fmaxf(x, -F16MAXV), F16MAXV);
                C[(size_t)m * ldc + n] = x;
            }
        }
    }
}

// y = shift + (1+scale) * layernorm(x), one row per block.
__global__ void ln_mod_rows(const float* __restrict__ x, const float* __restrict__ shift,
                            const float* __restrict__ scale, float* __restrict__ y, int n)
{
    int row = blockIdx.x, tid = threadIdx.x;
    const float* xr = x + (size_t)row * n;
    float* yr = y + (size_t)row * n;
    __shared__ float red[256];
    float s = 0.f;
    for (int c = tid; c < n; c += 256) s += xr[c];
    red[tid] = s; __syncthreads();
    for (int st = 128; st > 0; st >>= 1) { if (tid < st) red[tid] += red[tid + st]; __syncthreads(); }
    float mean = red[0] / n; __syncthreads();
    float v = 0.f;
    for (int c = tid; c < n; c += 256) { float d = xr[c] - mean; v += d * d; }
    red[tid] = v; __syncthreads();
    for (int st = 128; st > 0; st >>= 1) { if (tid < st) red[tid] += red[tid + st]; __syncthreads(); }
    float inv = rsqrtf(red[0] / n + 1e-6f);
    for (int c = tid; c < n; c += 256)
        yr[c] = shift[c] + (1.0f + scale[c]) * ((xr[c] - mean) * inv);
}

__global__ void rmsnorm_rows(const float* __restrict__ x, const float* __restrict__ scale,
                             float* __restrict__ y, int n)
{
    int row = blockIdx.x, tid = threadIdx.x;
    const float* xr = x + (size_t)row * n;
    float* yr = y + (size_t)row * n;
    __shared__ float red[256];
    float s = 0.f;
    for (int c = tid; c < n; c += 256) s += xr[c] * xr[c];
    red[tid] = s; __syncthreads();
    for (int st = 128; st > 0; st >>= 1) { if (tid < st) red[tid] += red[tid + st]; __syncthreads(); }
    float inv = rsqrtf(red[0] / n + 1e-6f);
    for (int c = tid; c < n; c += 256) yr[c] = xr[c] * inv * scale[c];
}

__global__ void softmax_rows(float* __restrict__ S, int n)
{
    long long row = blockIdx.x;
    int tid = threadIdx.x;
    float* r = S + row * n;
    __shared__ float red[256];
    float m = -3.0e38f;
    for (int c = tid; c < n; c += 256) m = fmaxf(m, r[c]);
    red[tid] = m; __syncthreads();
    for (int st = 128; st > 0; st >>= 1) { if (tid < st) red[tid] = fmaxf(red[tid], red[tid + st]); __syncthreads(); }
    float mx = red[0]; __syncthreads();
    float s = 0.f;
    for (int c = tid; c < n; c += 256) { float e = __expf(r[c] - mx); r[c] = e; s += e; }
    red[tid] = s; __syncthreads();
    for (int st = 128; st > 0; st >>= 1) { if (tid < st) red[tid] += red[tid + st]; __syncthreads(); }
    float inv = 1.0f / red[0];
    for (int c = tid; c < n; c += 256) r[c] *= inv;
}

// Split heads + rmsnorm(q,k) + rope; write Q,K,V as [H][token][128].
// qkv row layout = [3][HEADS][128]; grid (L, HEADS), 128 threads.
__global__ void qkv_post(const float* __restrict__ qkv, int stride,
                         const float* __restrict__ qs, const float* __restrict__ ks,
                         const float* __restrict__ cosb, const float* __restrict__ sinb,
                         int tokOff, float* __restrict__ Q, float* __restrict__ K,
                         float* __restrict__ V)
{
    int l = blockIdx.x, h = blockIdx.y, d = threadIdx.x;
    const float* row = qkv + (size_t)l * stride + h * HEAD_D;
    float q = row[d], k = row[HIDDEN + d], v = row[2 * HIDDEN + d];
    __shared__ float red[128], qa[128], ka[128];
    red[d] = q * q; __syncthreads();
    for (int st = 64; st > 0; st >>= 1) { if (d < st) red[d] += red[d + st]; __syncthreads(); }
    float qr = rsqrtf(red[0] / 128.0f + 1e-6f); __syncthreads();
    red[d] = k * k; __syncthreads();
    for (int st = 64; st > 0; st >>= 1) { if (d < st) red[d] += red[d + st]; __syncthreads(); }
    float kr = rsqrtf(red[0] / 128.0f + 1e-6f);
    qa[d] = q * qr * qs[d];
    ka[d] = k * kr * ks[d];
    __syncthreads();
    int tok = tokOff + l;
    size_t base = ((size_t)h * LALL + tok) * HEAD_D;
    V[base + d] = v;
    if (d < 64) {
        float c = cosb[tok * 64 + d], s = sinb[tok * 64 + d];
        float q0 = qa[2 * d], q1 = qa[2 * d + 1];
        Q[base + 2 * d]     = c * q0 - s * q1;
        Q[base + 2 * d + 1] = s * q0 + c * q1;
        float k0 = ka[2 * d], k1 = ka[2 * d + 1];
        K[base + 2 * d]     = c * k0 - s * k1;
        K[base + 2 * d + 1] = s * k0 + c * k1;
    }
}

// rope cos/sin tables for concatenated (txt, img) ids; axes dims (16,56,56).
__global__ void build_cs(const int* __restrict__ txt_ids, const int* __restrict__ img_ids,
                         float* __restrict__ cosb, float* __restrict__ sinb)
{
    int tok = blockIdx.x, p = threadIdx.x;   // p = 0..63 rotation pair
    int a, j; float dim;
    if (p < 8)       { a = 0; j = p;      dim = 16.f; }
    else if (p < 36) { a = 1; j = p - 8;  dim = 56.f; }
    else             { a = 2; j = p - 36; dim = 56.f; }
    int pos = (tok < LTXT) ? txt_ids[tok * 3 + a] : img_ids[(tok - LTXT) * 3 + a];
    float omega = __expf(-logf(10000.0f) * (2.0f * j) / dim);
    float arg = (float)pos * omega;
    cosb[tok * 64 + p] = cosf(arg);
    sinb[tok * 64 + p] = sinf(arg);
}

// approximator input: [17,64] = concat([dt(16), dg(16)] broadcast, mi(32))
__global__ void build_modin(const float* __restrict__ ts, const float* __restrict__ gd,
                            float* __restrict__ out)
{
    int idx = blockIdx.x * blockDim.x + threadIdx.x;
    if (idx >= MODL * 64) return;
    int r = idx >> 6, c = idx & 63;
    float t, half; int j; bool isCos;
    if (c < 32) {                                  // tg = [dt, dg]
        t = ((c < 16) ? ts[0] : gd[0]) * 1000.0f;
        j = c & 7; half = 8.0f; isCos = ((c & 15) < 8);
    } else {                                       // mi = emb(arange(17), 32)
        t = (float)r * 1000.0f;
        int cc = c - 32; j = cc & 15; half = 16.0f; isCos = (cc < 16);
    }
    float freq = __expf(-logf(10000.0f) * (float)j / half);
    float a = t * freq;
    out[idx] = isCos ? cosf(a) : sinf(a);
}

// cat[:, 3072:] = gelu(H[:, 9216:])   (single block concat(attn, gelu(mlp)))
__global__ void gelu_cat(const float* __restrict__ H, float* __restrict__ cat)
{
    long long total = (long long)LALL * MLPH;
    for (long long idx = blockIdx.x * 256LL + threadIdx.x; idx < total;
         idx += (long long)gridDim.x * 256LL) {
        long long m = idx / MLPH, j = idx % MLPH;
        cat[(size_t)m * 15360 + HIDDEN + j] = gelu_tanh_f(H[(size_t)m * 21504 + 3 * HIDDEN + j]);
    }
}

extern "C" void kernel_launch(void* const* d_in, const int* in_sizes, int n_in,
                              void* d_out, int out_size, void* d_ws, size_t ws_size,
                              hipStream_t stream)
{
    (void)in_sizes; (void)n_in; (void)out_size; (void)ws_size;

    // params flattened jax-tree style: nested dicts by sorted key, lists in order.
    enum {
        I_APPR_IN_B = 0, I_APPR_IN_W = 1,
        I_APPR_L1_B0 = 2,  I_APPR_L1_W0 = 7, I_APPR_L2_B0 = 12, I_APPR_L2_W0 = 17,
        I_APPR_NORM0 = 22, I_APPR_OUT_B = 27, I_APPR_OUT_W = 28,
        I_D_IMG_KS = 29, I_D_IMG_MLP1_B = 30, I_D_IMG_MLP1_W = 31, I_D_IMG_MLP2_B = 32,
        I_D_IMG_MLP2_W = 33, I_D_IMG_PROJ_B = 34, I_D_IMG_PROJ_W = 35, I_D_IMG_QKV_B = 36,
        I_D_IMG_QKV_W = 37, I_D_IMG_QS = 38,
        I_D_TXT_KS = 39, I_D_TXT_MLP1_B = 40, I_D_TXT_MLP1_W = 41, I_D_TXT_MLP2_B = 42,
        I_D_TXT_MLP2_W = 43, I_D_TXT_PROJ_B = 44, I_D_TXT_PROJ_W = 45, I_D_TXT_QKV_B = 46,
        I_D_TXT_QKV_W = 47, I_D_TXT_QS = 48,
        I_FINAL_B = 49, I_FINAL_W = 50, I_IMG_IN_B = 51, I_IMG_IN_W = 52,
        I_S_KS = 53, I_S_L1_B = 54, I_S_L1_W = 55, I_S_L2_B = 56, I_S_L2_W = 57, I_S_QS = 58,
        I_TXT_IN_B = 59, I_TXT_IN_W = 60,
        I_IMG = 61, I_TXT = 62, I_TS = 63, I_GUID = 64, I_IMG_IDS = 65, I_TXT_IDS = 66
    };
    auto p = [&](int i) { return (const float*)d_in[i]; };
    const float* img      = p(I_IMG);
    const float* txt      = p(I_TXT);
    const float* ts       = p(I_TS);
    const float* gd       = p(I_GUID);
    const int*   img_ids  = (const int*)d_in[I_IMG_IDS];
    const int*   txt_ids  = (const int*)d_in[I_TXT_IDS];
    float* out = (float*)d_out;

    // ---- workspace layout (fp32) -----------------------------------------
    float* W = (float*)d_ws;
    size_t off = 0;
    auto alloc = [&](size_t n) { float* q = W + off; off += n; return q; };
    float* modIn = alloc((size_t)MODL * 64);
    float* apprX = alloc((size_t)MODL * APPRH);
    float* apprN = alloc((size_t)MODL * APPRH);
    float* apprT = alloc((size_t)MODL * APPRH);
    float* mod   = alloc((size_t)MODL * HIDDEN);
    float* cosb  = alloc((size_t)LALL * 64);
    float* sinb  = alloc((size_t)LALL * 64);
    float* X     = alloc((size_t)LALL * HIDDEN);        // rows 0..511 txt, 512.. img
    float* XM    = alloc((size_t)LALL * HIDDEN);
    float* Hbuf  = alloc((size_t)LALL * 21504);
    float* Qb    = alloc((size_t)HEADS * LALL * HEAD_D);
    float* Kb    = alloc((size_t)HEADS * LALL * HEAD_D);
    float* Vb    = alloc((size_t)HEADS * LALL * HEAD_D);
    float* Sb    = alloc((size_t)HEADS * LALL * LALL);
    float* cat   = alloc((size_t)LALL * 15360);

    auto gemm = [&](const float* A, int lda, long long sA,
                    const float* Bm, int ldb, long long sB, int bNN,
                    float* C, int ldc, long long sC,
                    int M, int N, int K, int batch,
                    const float* bias, float alpha, int act,
                    const float* res, const float* gate, int clampOut) {
        dim3 grid((N + 63) / 64, (M + 127) / 128, batch);
        gemm_wmma<<<grid, 256, 0, stream>>>(A, lda, sA, Bm, ldb, sB, bNN,
                                            C, ldc, sC, M, N, K,
                                            bias, alpha, act, res, gate, clampOut);
    };
    auto mrow = [&](int r) { return mod + (size_t)r * HIDDEN; };

    // ---- approximator -> modulation vectors ------------------------------
    build_modin<<<(MODL * 64 + 255) / 256, 256, 0, stream>>>(ts, gd, modIn);
    gemm(modIn, 64, 0, p(I_APPR_IN_W), 64, 0, 0, apprX, APPRH, 0,
         MODL, APPRH, 64, 1, p(I_APPR_IN_B), 1.f, 0, nullptr, nullptr, 0);
    for (int i = 0; i < 5; ++i) {
        rmsnorm_rows<<<MODL, 256, 0, stream>>>(apprX, p(I_APPR_NORM0 + i), apprN, APPRH);
        gemm(apprN, APPRH, 0, p(I_APPR_L1_W0 + i), APPRH, 0, 0, apprT, APPRH, 0,
             MODL, APPRH, APPRH, 1, p(I_APPR_L1_B0 + i), 1.f, 2, nullptr, nullptr, 0);
        gemm(apprT, APPRH, 0, p(I_APPR_L2_W0 + i), APPRH, 0, 0, apprX, APPRH, 0,
             MODL, APPRH, APPRH, 1, p(I_APPR_L2_B0 + i), 1.f, 0, apprX, nullptr, 0);
    }
    gemm(apprX, APPRH, 0, p(I_APPR_OUT_W), APPRH, 0, 0, mod, HIDDEN, 0,
         MODL, HIDDEN, APPRH, 1, p(I_APPR_OUT_B), 1.f, 0, nullptr, nullptr, 0);

    // ---- input projections + rope tables ---------------------------------
    float* Xtxt = X;
    float* Ximg = X + (size_t)LTXT * HIDDEN;
    gemm(txt, CTXD, 0, p(I_TXT_IN_W), CTXD, 0, 0, Xtxt, HIDDEN, 0,
         LTXT, HIDDEN, CTXD, 1, p(I_TXT_IN_B), 1.f, 0, nullptr, nullptr, 0);
    gemm(img, INCH, 0, p(I_IMG_IN_W), INCH, 0, 0, Ximg, HIDDEN, 0,
         LIMG, HIDDEN, INCH, 1, p(I_IMG_IN_B), 1.f, 0, nullptr, nullptr, 0);
    build_cs<<<LALL, 64, 0, stream>>>(txt_ids, img_ids, cosb, sinb);

    const long long HS = (long long)LALL * HEAD_D;
    const long long SS = (long long)LALL * LALL;
    const float ISQ = 0.08838834764831845f;   // 128^-0.5

    // ======================= double block =================================
    // mod rows: single 0..2 | im1 3..5 | im2 6..8 | tm1 9..11 | tm2 12..14 | final 15,16
    ln_mod_rows<<<LTXT, 256, 0, stream>>>(Xtxt, mrow(9), mrow(10), XM, HIDDEN);
    ln_mod_rows<<<LIMG, 256, 0, stream>>>(Ximg, mrow(3), mrow(4), XM + (size_t)LTXT * HIDDEN, HIDDEN);
    float* txtQKV = Hbuf;
    float* imgQKV = Hbuf + (size_t)LTXT * 3 * HIDDEN;
    gemm(XM, HIDDEN, 0, p(I_D_TXT_QKV_W), HIDDEN, 0, 0, txtQKV, 3 * HIDDEN, 0,
         LTXT, 3 * HIDDEN, HIDDEN, 1, p(I_D_TXT_QKV_B), 1.f, 0, nullptr, nullptr, 0);
    gemm(XM + (size_t)LTXT * HIDDEN, HIDDEN, 0, p(I_D_IMG_QKV_W), HIDDEN, 0, 0,
         imgQKV, 3 * HIDDEN, 0, LIMG, 3 * HIDDEN, HIDDEN, 1,
         p(I_D_IMG_QKV_B), 1.f, 0, nullptr, nullptr, 0);
    qkv_post<<<dim3(LTXT, HEADS), 128, 0, stream>>>(txtQKV, 3 * HIDDEN, p(I_D_TXT_QS),
                                                    p(I_D_TXT_KS), cosb, sinb, 0, Qb, Kb, Vb);
    qkv_post<<<dim3(LIMG, HEADS), 128, 0, stream>>>(imgQKV, 3 * HIDDEN, p(I_D_IMG_QS),
                                                    p(I_D_IMG_KS), cosb, sinb, LTXT, Qb, Kb, Vb);
    // attention: S = softmax(Q K^T / sqrt(d)); attn = S V  -> cat[:, :3072]
    gemm(Qb, HEAD_D, HS, Kb, HEAD_D, HS, 0, Sb, LALL, SS,
         LALL, LALL, HEAD_D, HEADS, nullptr, ISQ, 0, nullptr, nullptr, 0);
    softmax_rows<<<HEADS * LALL, 256, 0, stream>>>(Sb, LALL);
    gemm(Sb, LALL, SS, Vb, HEAD_D, HS, 1, cat, HIDDEN, (long long)HEAD_D,
         LALL, HEAD_D, LALL, HEADS, nullptr, 1.f, 0, nullptr, nullptr, 0);
    // proj + gated residuals
    gemm(cat, HIDDEN, 0, p(I_D_TXT_PROJ_W), HIDDEN, 0, 0, Xtxt, HIDDEN, 0,
         LTXT, HIDDEN, HIDDEN, 1, p(I_D_TXT_PROJ_B), 1.f, 0, Xtxt, mrow(11), 0);
    gemm(cat + (size_t)LTXT * HIDDEN, HIDDEN, 0, p(I_D_IMG_PROJ_W), HIDDEN, 0, 0,
         Ximg, HIDDEN, 0, LIMG, HIDDEN, HIDDEN, 1,
         p(I_D_IMG_PROJ_B), 1.f, 0, Ximg, mrow(5), 0);
    // img MLP
    ln_mod_rows<<<LIMG, 256, 0, stream>>>(Ximg, mrow(6), mrow(7), XM, HIDDEN);
    gemm(XM, HIDDEN, 0, p(I_D_IMG_MLP1_W), HIDDEN, 0, 0, Hbuf, MLPH, 0,
         LIMG, MLPH, HIDDEN, 1, p(I_D_IMG_MLP1_B), 1.f, 1, nullptr, nullptr, 0);
    gemm(Hbuf, MLPH, 0, p(I_D_IMG_MLP2_W), MLPH, 0, 0, Ximg, HIDDEN, 0,
         LIMG, HIDDEN, MLPH, 1, p(I_D_IMG_MLP2_B), 1.f, 0, Ximg, mrow(8), 0);
    // txt MLP (block output clipped to +-F16_MAX)
    ln_mod_rows<<<LTXT, 256, 0, stream>>>(Xtxt, mrow(12), mrow(13), XM, HIDDEN);
    gemm(XM, HIDDEN, 0, p(I_D_TXT_MLP1_W), HIDDEN, 0, 0, Hbuf, MLPH, 0,
         LTXT, MLPH, HIDDEN, 1, p(I_D_TXT_MLP1_B), 1.f, 1, nullptr, nullptr, 0);
    gemm(Hbuf, MLPH, 0, p(I_D_TXT_MLP2_W), MLPH, 0, 0, Xtxt, HIDDEN, 0,
         LTXT, HIDDEN, MLPH, 1, p(I_D_TXT_MLP2_B), 1.f, 0, Xtxt, mrow(14), 1);

    // ======================= single block =================================
    ln_mod_rows<<<LALL, 256, 0, stream>>>(X, mrow(0), mrow(1), XM, HIDDEN);
    gemm(XM, HIDDEN, 0, p(I_S_L1_W), HIDDEN, 0, 0, Hbuf, 21504, 0,
         LALL, 21504, HIDDEN, 1, p(I_S_L1_B), 1.f, 0, nullptr, nullptr, 0);
    qkv_post<<<dim3(LALL, HEADS), 128, 0, stream>>>(Hbuf, 21504, p(I_S_QS), p(I_S_KS),
                                                    cosb, sinb, 0, Qb, Kb, Vb);
    gemm(Qb, HEAD_D, HS, Kb, HEAD_D, HS, 0, Sb, LALL, SS,
         LALL, LALL, HEAD_D, HEADS, nullptr, ISQ, 0, nullptr, nullptr, 0);
    softmax_rows<<<HEADS * LALL, 256, 0, stream>>>(Sb, LALL);
    gemm(Sb, LALL, SS, Vb, HEAD_D, HS, 1, cat, 15360, (long long)HEAD_D,
         LALL, HEAD_D, LALL, HEADS, nullptr, 1.f, 0, nullptr, nullptr, 0);
    gelu_cat<<<2048, 256, 0, stream>>>(Hbuf, cat);
    gemm(cat, 15360, 0, p(I_S_L2_W), 15360, 0, 0, X, HIDDEN, 0,
         LALL, HIDDEN, 15360, 1, p(I_S_L2_B), 1.f, 0, X, mrow(2), 1);

    // ======================= final layer ==================================
    ln_mod_rows<<<LIMG, 256, 0, stream>>>(Ximg, mrow(15), mrow(16), XM, HIDDEN);
    gemm(XM, HIDDEN, 0, p(I_FINAL_W), HIDDEN, 0, 0, out, INCH, 0,
         LIMG, INCH, HIDDEN, 1, p(I_FINAL_B), 1.f, 0, nullptr, nullptr, 0);
}
